// VNN_ResnetPointnet_OP_80135499809362
// MI455X (gfx1250) — compile-verified, matrix operand order checked by reading, not run on hardware
//
#include <hip/hip_runtime.h>
#include <hip/hip_bf16.h>
#include <stdint.h>

typedef __attribute__((ext_vector_type(16))) _Float16 v16h;
typedef __attribute__((ext_vector_type(8)))  _Float16 v8h;
typedef __attribute__((ext_vector_type(8)))  float    v8f;
typedef __attribute__((ext_vector_type(4)))  int      v4i;

#define EPSV   1e-6f
#define KNN    20
#define NPTS   4096
#define BATCH  4
#define MROWS  (BATCH * 3 * NPTS)   // 49152 rows, layout row = (b*3+d)*NPTS + n

// CDNA5 async L2->LDS path (ASYNCcnt-tracked), guarded so the file compiles
// on toolchains that do not declare the builtins (falls back to sync staging).
#if defined(__has_builtin)
#  if __has_builtin(__builtin_amdgcn_global_load_async_to_lds_b128) && \
      __has_builtin(__builtin_amdgcn_s_wait_asynccnt)
#    define USE_ASYNC_LDS 1
#  endif
#endif
#ifndef USE_ASYNC_LDS
#  define USE_ASYNC_LDS 0
#endif

#if USE_ASYNC_LDS
__device__ __forceinline__ void async_copy16B(const _Float16* g, _Float16* l) {
  // signature (from clang diagnostic): (v4i AS1*, v4i AS3*, imm offset, imm cpol)
  // generic LDS addr low 32 bits == LDS offset (aperture rule, ISA 10.2)
  __builtin_amdgcn_global_load_async_to_lds_b128(
      (__attribute__((address_space(1))) v4i*)(uintptr_t)g,
      (__attribute__((address_space(3))) v4i*)(unsigned int)(uintptr_t)l,
      0, 0);
}
#endif

// ---------------------------------------------------------------------------
// 1. kNN: per batch, all 4096 points staged in LDS (48 KB), each thread keeps
//    a fully-unrolled sorted top-20 list in registers (static indices only).
// ---------------------------------------------------------------------------
__global__ void __launch_bounds__(256)
knn_kernel(const float* __restrict__ p, int* __restrict__ knn_idx) {
  __shared__ float sx[NPTS], sy[NPTS], sz[NPTS];
  const int b = blockIdx.y;
  const float* pb = p + (size_t)b * NPTS * 3;
  for (int i = threadIdx.x; i < NPTS; i += 256) {
    sx[i] = pb[i * 3 + 0]; sy[i] = pb[i * 3 + 1]; sz[i] = pb[i * 3 + 2];
  }
  __syncthreads();
  const int n = blockIdx.x * 256 + threadIdx.x;
  const float cx = sx[n], cy = sy[n], cz = sz[n];
  float bd[KNN]; int bi[KNN];
#pragma unroll
  for (int j = 0; j < KNN; ++j) { bd[j] = 3.4e38f; bi[j] = 0; }
  for (int m = 0; m < NPTS; ++m) {
    const float dx = sx[m] - cx, dy = sy[m] - cy, dz = sz[m] - cz;
    const float d = dx * dx + dy * dy + dz * dz;
    if (d < bd[KNN - 1]) {
      float cd = d; int ci = m;
#pragma unroll
      for (int j = 0; j < KNN; ++j) {
        if (cd < bd[j]) {
          float td = bd[j]; int ti = bi[j];
          bd[j] = cd; bi[j] = ci; cd = td; ci = ti;
        }
      }
    }
  }
  int* o = knn_idx + ((size_t)b * NPTS + n) * KNN;
#pragma unroll
  for (int j = 0; j < KNN; ++j) o[j] = bi[j];
}

// ---------------------------------------------------------------------------
// 2. Edge features + conv_pos (3 -> 128) + leaky(0.2) + mean over k (VALU;
//    K=3 inner dim is useless for WMMA). Thread = (point, 16-channel chunk).
// ---------------------------------------------------------------------------
__global__ void __launch_bounds__(256)
edge_conv_kernel(const float* __restrict__ p, const int* __restrict__ knn_idx,
                 const float* __restrict__ Wf, const float* __restrict__ Wd,
                 _Float16* __restrict__ out) {
  __shared__ float sWf[128 * 3], sWd[128 * 3];
  for (int i = threadIdx.x; i < 384; i += 256) { sWf[i] = Wf[i]; sWd[i] = Wd[i]; }
  __syncthreads();
  const int gid = blockIdx.x * 256 + threadIdx.x;
  const int point = gid >> 3;        // b*N + n
  const int o0 = (gid & 7) * 16;
  const int b = point / NPTS, n = point - b * NPTS;
  const float* pb = p + (size_t)b * NPTS * 3;
  const float cx = pb[n * 3], cy = pb[n * 3 + 1], cz = pb[n * 3 + 2];
  const int* nb = knn_idx + (size_t)point * KNN;
  float acc[16][3];
#pragma unroll
  for (int o = 0; o < 16; ++o) { acc[o][0] = 0.f; acc[o][1] = 0.f; acc[o][2] = 0.f; }
  for (int kk = 0; kk < KNN; ++kk) {
    const int m = nb[kk];
    const float nx = pb[m * 3], ny = pb[m * 3 + 1], nz = pb[m * 3 + 2];
    const float f0x = nx - cx, f0y = ny - cy, f0z = nz - cz;
    const float f1x = cx, f1y = cy, f1z = cz;
    const float f2x = ny * cz - nz * cy;   // cross(nbr, ctr)
    const float f2y = nz * cx - nx * cz;
    const float f2z = nx * cy - ny * cx;
#pragma unroll
    for (int o = 0; o < 16; ++o) {
      const int oo = o0 + o;
      const float a0 = sWf[oo * 3], a1 = sWf[oo * 3 + 1], a2 = sWf[oo * 3 + 2];
      const float b0 = sWd[oo * 3], b1 = sWd[oo * 3 + 1], b2 = sWd[oo * 3 + 2];
      const float pfx = a0 * f0x + a1 * f1x + a2 * f2x;
      const float pfy = a0 * f0y + a1 * f1y + a2 * f2y;
      const float pfz = a0 * f0z + a1 * f1z + a2 * f2z;
      const float dxv = b0 * f0x + b1 * f1x + b2 * f2x;
      const float dyv = b0 * f0y + b1 * f1y + b2 * f2y;
      const float dzv = b0 * f0z + b1 * f1z + b2 * f2z;
      const float dot = pfx * dxv + pfy * dyv + pfz * dzv;
      const float dsq = dxv * dxv + dyv * dyv + dzv * dzv;
      const float fac = dot / (dsq + EPSV);
      const bool pos = dot >= 0.f;
      const float rx = pos ? pfx : pfx - fac * dxv;
      const float ry = pos ? pfy : pfy - fac * dyv;
      const float rz = pos ? pfz : pfz - fac * dzv;
      acc[o][0] += 0.2f * pfx + 0.8f * rx;
      acc[o][1] += 0.2f * pfy + 0.8f * ry;
      acc[o][2] += 0.2f * pfz + 0.8f * rz;
    }
  }
  const float inv = 1.0f / (float)KNN;
#pragma unroll
  for (int o = 0; o < 16; ++o)
#pragma unroll
    for (int d = 0; d < 3; ++d) {
      const size_t r = (size_t)(b * 3 + d) * NPTS + n;
      out[r * 128 + o0 + o] = (_Float16)(acc[o][d] * inv);
    }
}

// ---------------------------------------------------------------------------
// 3. fp32 -> f16 weight pre-conversion (runs once per launch; weights ~1.7 MB)
// ---------------------------------------------------------------------------
__global__ void __launch_bounds__(256)
cvt_f16_kernel(const float* __restrict__ src, _Float16* __restrict__ dst, int n) {
  const int i = blockIdx.x * 256 + threadIdx.x;
  if (i < n) dst[i] = (_Float16)src[i];
}

// ---------------------------------------------------------------------------
// 4. WMMA GEMM: Out[M,N](f16) = A[M,K](f16) x W[N,K](f16)^T (+ Add[M,N])
//    Block 256 thr = 8 waves (4 row x 2 col); block tile 128x128;
//    wave tile 32x64 -> 8x v_wmma_f32_16x16x32_f16 per 32-k step.
//    Double-buffered LDS; async L2->LDS pipeline when available.
// ---------------------------------------------------------------------------
__global__ void __launch_bounds__(256)
gemm_wmma_kernel(const _Float16* __restrict__ A, const _Float16* __restrict__ W,
                 const _Float16* __restrict__ Add, _Float16* __restrict__ Out,
                 int K, int Nout) {
  __shared__ _Float16 sA[2][128 * 40];   // pitch 40 halfs (80 B, 16B-aligned frags)
  __shared__ _Float16 sW[2][128 * 40];
  const int tid  = threadIdx.x;
  const int wave = tid >> 5, lane = tid & 31;
  const int wr = wave >> 1, wc = wave & 1;        // 4x2 wave grid
  const int rowBase = blockIdx.x * 128;
  const int colBase = blockIdx.y * 128;
  const int lrow  = lane & 15;        // row/col inside a 16-tile for this lane
  const int kbase = (lane >> 4) * 8;  // ISA 16-bit A/B fragment k-interleave
  const int trow  = tid >> 1;         // staging: 2 threads per tile row
  const int thalf = (tid & 1) * 16;   // 16 halfs (32 B) per thread

  const _Float16* gA = A + (size_t)(rowBase + trow) * K + thalf;
  const _Float16* gW = W + (size_t)(colBase + trow) * K + thalf;
  _Float16* lA0 = &sA[0][trow * 40 + thalf];
  _Float16* lA1 = &sA[1][trow * 40 + thalf];
  _Float16* lW0 = &sW[0][trow * 40 + thalf];
  _Float16* lW1 = &sW[1][trow * 40 + thalf];

  v8f acc[2][4];
  const v8f vzero = {};
#pragma unroll
  for (int i = 0; i < 2; ++i)
#pragma unroll
    for (int j = 0; j < 4; ++j) acc[i][j] = vzero;

  auto load_tiles = [&](int buf, int k0) {
    _Float16* la = buf ? lA1 : lA0;
    _Float16* lw = buf ? lW1 : lW0;
#if USE_ASYNC_LDS
    async_copy16B(gA + k0,     la);
    async_copy16B(gA + k0 + 8, la + 8);
    async_copy16B(gW + k0,     lw);
    async_copy16B(gW + k0 + 8, lw + 8);
#else
    *(v8h*)(la)     = *(const v8h*)(gA + k0);
    *(v8h*)(la + 8) = *(const v8h*)(gA + k0 + 8);
    *(v8h*)(lw)     = *(const v8h*)(gW + k0);
    *(v8h*)(lw + 8) = *(const v8h*)(gW + k0 + 8);
#endif
  };

  const int steps = K >> 5;
  load_tiles(0, 0);
  for (int s = 0; s < steps; ++s) {
    const int buf = s & 1;
    const bool more = (s + 1) < steps;
    if (more) load_tiles(buf ^ 1, (s + 1) << 5);   // stream next tile in bkgnd
#if USE_ASYNC_LDS
    if (more) __builtin_amdgcn_s_wait_asynccnt(4); // only tile s must be done
    else      __builtin_amdgcn_s_wait_asynccnt(0);
#endif
    __syncthreads();

    union { v16h v; v8h h[2]; } af[2], bf[4];
#pragma unroll
    for (int tm = 0; tm < 2; ++tm) {
      const _Float16* pA = &sA[buf][(wr * 32 + tm * 16 + lrow) * 40];
      af[tm].h[0] = *(const v8h*)(pA + kbase);
      af[tm].h[1] = *(const v8h*)(pA + kbase + 16);
    }
#pragma unroll
    for (int tn = 0; tn < 4; ++tn) {
      const _Float16* pW = &sW[buf][(wc * 64 + tn * 16 + lrow) * 40];
      bf[tn].h[0] = *(const v8h*)(pW + kbase);
      bf[tn].h[1] = *(const v8h*)(pW + kbase + 16);
    }
#pragma unroll
    for (int tm = 0; tm < 2; ++tm)
#pragma unroll
      for (int tn = 0; tn < 4; ++tn)
        acc[tm][tn] = __builtin_amdgcn_wmma_f32_16x16x32_f16(
            false, af[tm].v, false, bf[tn].v, (short)0, acc[tm][tn], false, false);
    __syncthreads();   // protect buffer reuse two steps later
  }

  // D layout: lane holds col (lane&15); VGPR r -> row r + 8*(lane>>4)
  const int rowHalf = (lane >> 4) * 8;
#pragma unroll
  for (int tm = 0; tm < 2; ++tm)
#pragma unroll
    for (int tn = 0; tn < 4; ++tn) {
      const v8f accv = acc[tm][tn];
      const int gcol = colBase + wc * 64 + tn * 16 + lrow;
#pragma unroll
      for (int reg = 0; reg < 8; ++reg) {
        const int grow = rowBase + wr * 32 + tm * 16 + reg + rowHalf;
        float v = accv[reg];
        if (Add) v += (float)Add[(size_t)grow * Nout + gcol];
        Out[(size_t)grow * Nout + gcol] = (_Float16)v;
      }
    }
}

// ---------------------------------------------------------------------------
// 5. vn_leaky elementwise: dot/dsq across the 3 vector components.
// ---------------------------------------------------------------------------
__global__ void __launch_bounds__(256)
vn_leaky_kernel(const _Float16* __restrict__ X, const _Float16* __restrict__ D,
                _Float16* __restrict__ Out, int C, float slope) {
  const size_t idx = (size_t)blockIdx.x * 256 + threadIdx.x;   // over B*N*C
  if (idx >= (size_t)BATCH * NPTS * C) return;
  const int c = (int)(idx % C);
  const size_t bn = idx / C;
  const int n = (int)(bn % NPTS);
  const int b = (int)(bn / NPTS);
  const size_t r0 = ((size_t)(b * 3 + 0) * NPTS + n) * C + c;
  const size_t r1 = ((size_t)(b * 3 + 1) * NPTS + n) * C + c;
  const size_t r2 = ((size_t)(b * 3 + 2) * NPTS + n) * C + c;
  const float x0 = (float)X[r0], x1 = (float)X[r1], x2 = (float)X[r2];
  const float d0 = (float)D[r0], d1 = (float)D[r1], d2 = (float)D[r2];
  const float dot = x0 * d0 + x1 * d1 + x2 * d2;
  const float dsq = d0 * d0 + d1 * d1 + d2 * d2;
  const float fac = dot / (dsq + EPSV);
  const bool pos = dot >= 0.f;
  const float om = 1.f - slope;
  Out[r0] = (_Float16)(slope * x0 + om * (pos ? x0 : x0 - fac * d0));
  Out[r1] = (_Float16)(slope * x1 + om * (pos ? x1 : x1 - fac * d1));
  Out[r2] = (_Float16)(slope * x2 + om * (pos ? x2 : x2 - fac * d2));
}

// 6. Mean over n (C=128) -> fp32 pooled[b*3+d][128]
__global__ void __launch_bounds__(128)
meanpool_kernel(const _Float16* __restrict__ Y, float* __restrict__ pooled) {
  const int bd = blockIdx.x, c = threadIdx.x;
  const _Float16* base = Y + (size_t)bd * NPTS * 128 + c;
  float s = 0.f;
  for (int n = 0; n < NPTS; ++n) s += (float)base[(size_t)n * 128];
  pooled[bd * 128 + c] = s * (1.0f / NPTS);
}

// 7. concat [Y | broadcast(pooled)] -> 256-channel input for next resblock
__global__ void __launch_bounds__(256)
concat_kernel(const _Float16* __restrict__ Y, const float* __restrict__ pooled,
              _Float16* __restrict__ Xn) {
  const size_t idx = (size_t)blockIdx.x * 256 + threadIdx.x;  // MROWS*256
  const int c = (int)(idx & 255);
  const size_t r = idx >> 8;
  const int bd = (int)(r / NPTS);
  Xn[idx] = (c < 128) ? Y[r * 128 + c] : (_Float16)pooled[bd * 128 + (c - 128)];
}

// 8. head: vn_leaky(meanpool, actc_dir, 0.2) on [B,128,3] (fp32, tiny)
__global__ void __launch_bounds__(128)
head_leaky_kernel(const float* __restrict__ nm, const float* __restrict__ Wdir,
                  float* __restrict__ na) {
  __shared__ float xm[3][128];
  const int b = blockIdx.x, o = threadIdx.x;
  for (int d = 0; d < 3; ++d) xm[d][o] = nm[(b * 3 + d) * 128 + o];
  __syncthreads();
  float dv0 = 0.f, dv1 = 0.f, dv2 = 0.f;
  for (int h = 0; h < 128; ++h) {
    const float w = Wdir[o * 128 + h];
    dv0 += w * xm[0][h]; dv1 += w * xm[1][h]; dv2 += w * xm[2][h];
  }
  const float dot = xm[0][o] * dv0 + xm[1][o] * dv1 + xm[2][o] * dv2;
  const float dsq = dv0 * dv0 + dv1 * dv1 + dv2 * dv2;
  const float fac = dot / (dsq + EPSV);
  const bool pos = dot >= 0.f;
  const float dv[3] = {dv0, dv1, dv2};
  for (int d = 0; d < 3; ++d) {
    const float x = xm[d][o];
    na[(b * 3 + d) * 128 + o] = 0.2f * x + 0.8f * (pos ? x : x - fac * dv[d]);
  }
}

// 9. hypernet hidden layers (two relu MLP layers for both w and b paths)
__global__ void __launch_bounds__(256)
hyper_h_kernel(const float* __restrict__ z, const int* __restrict__ zidx,
               const float* __restrict__ hw0, const float* __restrict__ hw1,
               const float* __restrict__ hb0, const float* __restrict__ hb1,
               float* __restrict__ h1w, float* __restrict__ h1b) {
  __shared__ float zf[256], hbuf[256];
  const int b = blockIdx.x, t = threadIdx.x;
  zf[t] = z[(size_t)zidx[b] * 256 + t];
  __syncthreads();
  float s = 0.f;
  for (int c = 0; c < 256; ++c) s += zf[c] * hw0[t * 256 + c];
  hbuf[t] = fmaxf(s, 0.f);
  __syncthreads();
  float s1 = 0.f;
  for (int c = 0; c < 256; ++c) s1 += hbuf[c] * hw1[t * 256 + c];
  h1w[b * 256 + t] = fmaxf(s1, 0.f);
  __syncthreads();
  float sb = 0.f;
  for (int c = 0; c < 256; ++c) sb += zf[c] * hb0[t * 256 + c];
  hbuf[t] = fmaxf(sb, 0.f);
  __syncthreads();
  float sb1 = 0.f;
  for (int c = 0; c < 256; ++c) sb1 += hbuf[c] * hb1[t * 256 + c];
  h1b[b * 256 + t] = fmaxf(sb1, 0.f);
}

// 10. hypernet final projections
__global__ void __launch_bounds__(256)
hyper_w_kernel(const float* __restrict__ h1w, const float* __restrict__ hw2,
               float* __restrict__ wmat) {
  const int idx = blockIdx.x * 256 + threadIdx.x;  // 4*16384
  const int b = idx >> 14, j = idx & 16383;
  const float* h = h1w + b * 256;
  const float* w = hw2 + (size_t)j * 256;
  float s = 0.f;
  for (int c = 0; c < 256; ++c) s += h[c] * w[c];
  wmat[idx] = s;
}

__global__ void __launch_bounds__(256)
hyper_b_kernel(const float* __restrict__ h1b, const float* __restrict__ hb2,
               float* __restrict__ bvec) {
  const int idx = blockIdx.x * 256 + threadIdx.x;  // 512
  const int b = idx >> 7, j = idx & 127;
  const float* h = h1b + b * 256;
  const float* w = hb2 + (size_t)j * 256;
  float s = 0.f;
  for (int c = 0; c < 256; ++c) s += h[c] * w[c];
  bvec[idx] = s;
}

// 11. c[b,cc,d] = sum_h net[b,d,h] * w[b,cc,h] + bvec[b,cc]
__global__ void __launch_bounds__(256)
final_out_kernel(const float* __restrict__ na, const float* __restrict__ wmat,
                 const float* __restrict__ bvec, float* __restrict__ out) {
  const int idx = blockIdx.x * 256 + threadIdx.x;  // b*128+cc, 512 total
  const int b = idx >> 7, cc = idx & 127;
  const float* w = wmat + (size_t)b * 16384 + cc * 128;
  const float bb = bvec[idx];
  for (int d = 0; d < 3; ++d) {
    const float* x = na + (b * 3 + d) * 128;
    float s = 0.f;
    for (int h = 0; h < 128; ++h) s += x[h] * w[h];
    out[(size_t)idx * 3 + d] = s + bb;
  }
}

// ---------------------------------------------------------------------------
extern "C" void kernel_launch(void* const* d_in, const int* in_sizes, int n_in,
                              void* d_out, int out_size, void* d_ws, size_t ws_size,
                              hipStream_t stream) {
  (void)in_sizes; (void)n_in; (void)out_size; (void)ws_size;
  const float* p        = (const float*)d_in[0];
  const int*   zidx     = (const int*)d_in[1];
  const float* z        = (const float*)d_in[2];
  const float* hw0      = (const float*)d_in[3];
  const float* hw1      = (const float*)d_in[4];
  const float* hw2      = (const float*)d_in[5];
  const float* hb0      = (const float*)d_in[6];
  const float* hb1      = (const float*)d_in[7];
  const float* hb2      = (const float*)d_in[8];
  const float* cpf      = (const float*)d_in[9];
  const float* cpd      = (const float*)d_in[10];
  const float* fc_pos_w = (const float*)d_in[11];
  const float* blk_fc0  = (const float*)d_in[12];  // [5,128,256]
  const float* blk_fc1  = (const float*)d_in[13];  // [5,128,128]
  const float* blk_a0   = (const float*)d_in[14];  // [5,256,256]
  const float* blk_a1   = (const float*)d_in[15];  // [5,128,128]
  const float* blk_sc   = (const float*)d_in[16];  // [5,128,256]
  const float* actc     = (const float*)d_in[17];  // [128,128]

  // workspace carve-up (256-byte aligned slabs)
  char* ws = (char*)d_ws;
  size_t off = 0;
  auto carve = [&](size_t bytes) { void* r = ws + off; off += (bytes + 255) & ~(size_t)255; return r; };
  int*       knn_idx = (int*)      carve((size_t)BATCH * NPTS * KNN * 4);   // 1.25 MB
  _Float16*  bufA    = (_Float16*) carve((size_t)MROWS * 256 * 2);          // 25 MB  X (256ch)
  _Float16*  bufB    = (_Float16*) carve((size_t)MROWS * 256 * 2);          // 25 MB  leaky out
  _Float16*  bufC    = (_Float16*) carve((size_t)MROWS * 128 * 2);          // 12.6 MB net / Y
  _Float16*  bufD    = (_Float16*) carve((size_t)MROWS * 128 * 2);          // 12.6 MB D2 / dx
  _Float16*  bufF    = (_Float16*) carve((size_t)MROWS * 256 * 2);          // 25 MB  dir matrix
  // f16 weight slabs
  _Float16*  w_fcpos = (_Float16*) carve((size_t)256 * 128 * 2);
  _Float16*  w_a0    = (_Float16*) carve((size_t)5 * 256 * 256 * 2);
  _Float16*  w_fc0   = (_Float16*) carve((size_t)5 * 128 * 256 * 2);
  _Float16*  w_a1    = (_Float16*) carve((size_t)5 * 128 * 128 * 2);
  _Float16*  w_fc1   = (_Float16*) carve((size_t)5 * 128 * 128 * 2);
  _Float16*  w_sc    = (_Float16*) carve((size_t)5 * 128 * 256 * 2);
  float*     pooled  = (float*)    carve((size_t)BATCH * 3 * 128 * 4);
  float*     netmean = (float*)    carve((size_t)BATCH * 3 * 128 * 4);
  float*     netact  = (float*)    carve((size_t)BATCH * 3 * 128 * 4);
  float*     h1w     = (float*)    carve((size_t)BATCH * 256 * 4);
  float*     h1b     = (float*)    carve((size_t)BATCH * 256 * 4);
  float*     wmat    = (float*)    carve((size_t)BATCH * 16384 * 4);
  float*     bvec    = (float*)    carve((size_t)BATCH * 128 * 4);

  auto cvt = [&](const float* src, _Float16* dst, int n) {
    cvt_f16_kernel<<<(n + 255) / 256, 256, 0, stream>>>(src, dst, n);
  };
  auto gemm = [&](const _Float16* A, const _Float16* W, const _Float16* Add,
                  _Float16* Out, int Kd, int Nd) {
    dim3 grid(MROWS / 128, Nd / 128);
    gemm_wmma_kernel<<<grid, 256, 0, stream>>>(A, W, Add, Out, Kd, Nd);
  };

  // 0. pre-convert weights to f16 (reused 384x per GEMM; removes cvt from hot loop)
  cvt(fc_pos_w, w_fcpos, 256 * 128);
  cvt(blk_a0,  w_a0,  5 * 256 * 256);
  cvt(blk_fc0, w_fc0, 5 * 128 * 256);
  cvt(blk_a1,  w_a1,  5 * 128 * 128);
  cvt(blk_fc1, w_fc1, 5 * 128 * 128);
  cvt(blk_sc,  w_sc,  5 * 128 * 256);

  // 1. kNN + edge conv -> net1 [M,128] in bufC
  knn_kernel<<<dim3(NPTS / 256, BATCH), 256, 0, stream>>>(p, knn_idx);
  edge_conv_kernel<<<(BATCH * NPTS * 8) / 256, 256, 0, stream>>>(p, knn_idx, cpf, cpd, bufC);

  // 2. fc_pos: [M,128] -> [M,256]
  gemm(bufC, w_fcpos, nullptr, bufA, 128, 256);

  // 3. five resblocks
  const int g256 = (BATCH * NPTS * 256) / 256;   // leaky grid for C=256
  const int g128 = (BATCH * NPTS * 128) / 256;   // leaky grid for C=128
  for (int i = 0; i < 5; ++i) {
    const _Float16* a0 = w_a0  + (size_t)i * 256 * 256;
    const _Float16* f0 = w_fc0 + (size_t)i * 128 * 256;
    const _Float16* a1 = w_a1  + (size_t)i * 128 * 128;
    const _Float16* f1 = w_fc1 + (size_t)i * 128 * 128;
    const _Float16* sc = w_sc  + (size_t)i * 128 * 256;

    gemm(bufA, a0, nullptr, bufF, 256, 256);                                  // D = X*a0^T
    vn_leaky_kernel<<<g256, 256, 0, stream>>>(bufA, bufF, bufB, 256, 0.f);    // leaky0
    gemm(bufB, f0, nullptr, bufC, 256, 128);                                  // net
    gemm(bufC, a1, nullptr, bufD, 128, 128);                                  // D2
    vn_leaky_kernel<<<g128, 256, 0, stream>>>(bufC, bufD, bufB, 128, 0.f);    // leaky1
    gemm(bufB, f1, nullptr, bufD, 128, 128);                                  // dx
    gemm(bufA, sc, bufD, bufC, 256, 128);                                     // Y = X*sc^T + dx
    if (i < 4) {
      meanpool_kernel<<<BATCH * 3, 128, 0, stream>>>(bufC, pooled);
      concat_kernel<<<MROWS, 256, 0, stream>>>(bufC, pooled, bufA);           // next X
    }
  }

  // 4. head: mean over n, vn_leaky(actc_dir, 0.2)
  meanpool_kernel<<<BATCH * 3, 128, 0, stream>>>(bufC, netmean);
  head_leaky_kernel<<<BATCH, 128, 0, stream>>>(netmean, actc, netact);

  // 5. hypernet + final einsum
  hyper_h_kernel<<<BATCH, 256, 0, stream>>>(z, zidx, hw0, hw1, hb0, hb1, h1w, h1b);
  hyper_w_kernel<<<(BATCH * 16384) / 256, 256, 0, stream>>>(h1w, hw2, wmat);
  hyper_b_kernel<<<(BATCH * 128) / 256, 256, 0, stream>>>(h1b, hb2, bvec);
  final_out_kernel<<<(BATCH * 128) / 256, 256, 0, stream>>>(netact, wmat, bvec, (float*)d_out);
}
